// LanguageModel_60206851555776
// MI455X (gfx1250) — compile-verified, matrix operand order checked by reading, not run on hardware
//
#include <hip/hip_runtime.h>

#define V_ 32000
#define E_ 65
#define H_ 65
#define B_ 8
#define T_ 512
#define KP 96                 // K padded to 3x32 for wmma 16x16x32
#define M_ (B_*T_)            // 4096 rows

typedef __attribute__((ext_vector_type(16))) _Float16 v16h;
typedef __attribute__((ext_vector_type(8)))  float    v8f;

__device__ __forceinline__ float sig_(float x){ return 1.0f/(1.0f+expf(-x)); }

// Decode fragment-order linear index i = ((tile*3+ks)*32+lane)*16+e  ->  (tile,row-in-tile,k)
// Per ISA 16-bit A/B layout: lane<16 holds K = k0+{0..7, 16..23}; lane>=16 holds K = k0+{8..15, 24..31}.
__device__ __forceinline__ void frag_decode(int i, int& tile, int& r16, int& k){
    int e    = i & 15;
    int lane = (i >> 4) & 31;
    int rem  = i >> 9;
    int ks   = rem % 3;
    tile     = rem / 3;
    r16      = lane & 15;
    k        = ks*32 + ((lane >= 16) ? 8 : 0) + e + ((e >= 8) ? 8 : 0);
}

// ---------------- embedding gather ----------------
__global__ void k_embed(const int* __restrict__ ids, const float* __restrict__ emb,
                        float* __restrict__ xe){
    int i = blockIdx.x*blockDim.x + threadIdx.x;
    if(i >= M_*E_) return;
    int e = i % E_, bt = i / E_;
    xe[i] = emb[(size_t)ids[bt]*E_ + e];
}

// ---------------- sequential LSTM scan (one workgroup) ----------------
__global__ void k_lstm(const float* __restrict__ xe,
                       const float* __restrict__ w_ih, const float* __restrict__ w_hh,
                       const float* __restrict__ b_ih, const float* __restrict__ b_hh,
                       const float* __restrict__ h0, const float* __restrict__ c0,
                       float* __restrict__ enc, float* __restrict__ hT, float* __restrict__ cT){
    __shared__ float hbuf[2][B_*H_];
    __shared__ float cbuf[2][B_*H_];
    for(int idx=threadIdx.x; idx<B_*H_; idx+=blockDim.x){
        hbuf[0][idx] = h0[idx]; cbuf[0][idx] = c0[idx];
    }
    __syncthreads();
    for(int t=0; t<T_; ++t){
        int src = t & 1, dst = src ^ 1;
        for(int idx=threadIdx.x; idx<B_*H_; idx+=blockDim.x){
            int b = idx / H_, h = idx % H_;
            const float* xr = xe + ((size_t)(b*T_+t))*E_;
            const float* hp = &hbuf[src][b*H_];
            float g[4];
            #pragma unroll
            for(int gi=0; gi<4; ++gi){
                int row = gi*H_ + h;
                float acc = b_ih[row] + b_hh[row];
                const float* wi = w_ih + (size_t)row*E_;
                const float* wh = w_hh + (size_t)row*H_;
                for(int k=0;k<E_;++k) acc += wi[k]*xr[k];
                for(int k=0;k<H_;++k) acc += wh[k]*hp[k];
                g[gi] = acc;
            }
            float ii=sig_(g[0]), ff=sig_(g[1]), gg=tanhf(g[2]), oo=sig_(g[3]);
            float cc = ff*cbuf[src][idx] + ii*gg;
            float hh = oo*tanhf(cc);
            cbuf[dst][idx]=cc; hbuf[dst][idx]=hh;
            enc[((size_t)(b*T_+t))*H_ + h] = hh;
        }
        __syncthreads();
    }
    const int fin = T_ & 1;   // T even -> final state in buffer 0
    for(int idx=threadIdx.x; idx<B_*H_; idx+=blockDim.x){
        hT[idx]=hbuf[fin][idx]; cT[idx]=cbuf[fin][idx];
    }
}

// ---------------- keys = W(enc)+wb, queries = U(enc)+ub ----------------
__global__ void k_kq(const float* __restrict__ enc,
                     const float* __restrict__ u_w, const float* __restrict__ u_b,
                     const float* __restrict__ w_w, const float* __restrict__ w_b,
                     float* __restrict__ keys, float* __restrict__ qrs){
    int i = blockIdx.x*blockDim.x + threadIdx.x;
    if(i >= M_*H_) return;
    int h = i % H_, bt = i / H_;
    const float* er = enc + (size_t)bt*H_;
    const float* wr = w_w + (size_t)h*H_;
    const float* ur = u_w + (size_t)h*H_;
    float ak = w_b[h], aq = u_b[h];
    for(int k=0;k<H_;++k){ float e=er[k]; ak += wr[k]*e; aq += ur[k]*e; }
    keys[i]=ak; qrs[i]=aq;
}

// ---------------- per-(b,j) attention scores (two values, not T^2) ----------------
__global__ void k_scores(const float* __restrict__ keys, const float* __restrict__ qrs,
                         const float* __restrict__ v_w, const float* __restrict__ v_b,
                         float* __restrict__ s_on, float* __restrict__ s_off){
    int bt = blockIdx.x*blockDim.x + threadIdx.x;
    if(bt >= M_) return;
    const float* kr = keys + (size_t)bt*H_;
    const float* qr = qrs  + (size_t)bt*H_;
    float on = v_b[0], off = v_b[0];
    for(int h=0;h<H_;++h){ float v=v_w[h]; on += v*tanhf(kr[h]+qr[h]); off += v*tanhf(kr[h]); }
    s_on[bt]=on; s_off[bt]=off;
}

// ---------------- per-batch max, exp, scalar prefix sums ----------------
__global__ void k_smax(float* __restrict__ s_on, float* __restrict__ s_off,
                       float* __restrict__ S_on, float* __restrict__ S_off){
    __shared__ float red[T_];
    int b = blockIdx.x, t = threadIdx.x;
    float a = s_on[b*T_+t], o = s_off[b*T_+t];
    red[t] = fmaxf(a,o);
    __syncthreads();
    for(int s=T_/2; s>0; s>>=1){ if(t<s) red[t]=fmaxf(red[t],red[t+s]); __syncthreads(); }
    float m = red[0];
    s_on[b*T_+t]  = expf(a-m);
    s_off[b*T_+t] = expf(o-m);
    __syncthreads();
    if(t==0){
        float ca=0.f, co=0.f;
        for(int j=0;j<T_;++j){
            ca += s_on[b*T_+j];  S_on[b*T_+j]=ca;
            co += s_off[b*T_+j]; S_off[b*T_+j]=co;
        }
    }
}

// ---------------- vector prefix sums: P_on/P_off[b,t,h] ----------------
__global__ void k_prefix(const float* __restrict__ e_on, const float* __restrict__ e_off,
                         const float* __restrict__ qrs,
                         float* __restrict__ P_on, float* __restrict__ P_off){
    int idx = blockIdx.x*blockDim.x + threadIdx.x;
    if(idx >= B_*H_) return;
    int b = idx / H_, h = idx % H_;
    float aon=0.f, aoff=0.f;
    for(int t=0;t<T_;++t){
        size_t p = (size_t)(b*T_+t);
        float q = qrs[p*H_+h];
        aon  += e_on[p]*q;  P_on[p*H_+h]=aon;
        aoff += e_off[p]*q; P_off[p*H_+h]=aoff;
    }
}

// ---------------- context via prefix identity ----------------
__global__ void k_ctx(const float* __restrict__ P_on, const float* __restrict__ P_off,
                      const float* __restrict__ S_on, const float* __restrict__ S_off,
                      float* __restrict__ ctx){
    int i = blockIdx.x*blockDim.x + threadIdx.x;
    if(i >= M_*H_) return;
    int h = i % H_, bt = i / H_, b = bt / T_;
    float Z    = S_on[bt] + (S_off[b*T_+T_-1] - S_off[bt]);
    float Ptot = P_off[((size_t)(b*T_+T_-1))*H_ + h];
    ctx[i] = (P_on[i] + Ptot - P_off[i]) / Z;
}

// ---- z = enc@ohW.T + ctx@ozW.T + biases, written f16 directly in WMMA A-fragment order ----
__global__ void k_zf(const float* __restrict__ enc, const float* __restrict__ ctx,
                     const float* __restrict__ oh_w, const float* __restrict__ oh_b,
                     const float* __restrict__ oz_w, const float* __restrict__ oz_b,
                     _Float16* __restrict__ zpf){
    int i = blockIdx.x*blockDim.x + threadIdx.x;
    if(i >= M_*KP) return;
    int mtile, r16, k;
    frag_decode(i, mtile, r16, k);
    int row = mtile*16 + r16;
    float r = 0.f;
    if(k < H_){
        const float* er  = enc  + (size_t)row*H_;
        const float* cr  = ctx  + (size_t)row*H_;
        const float* whr = oh_w + (size_t)k*H_;
        const float* wzr = oz_w + (size_t)k*H_;
        r = oh_b[k] + oz_b[k];
        for(int q=0;q<H_;++q) r += er[q]*whr[q] + cr[q]*wzr[q];
    }
    zpf[i] = (_Float16)r;
}

// ---- ov_w -> f16 in WMMA B-fragment order: [ntile16][kstep][lane][16] ----
__global__ void k_ovf(const float* __restrict__ ov_w, _Float16* __restrict__ ovf){
    int i = blockIdx.x*blockDim.x + threadIdx.x;
    if(i >= V_*KP) return;
    int ntile, c16, k;
    frag_decode(i, ntile, c16, k);
    int col = ntile*16 + c16;
    ovf[i] = (_Float16)((k < H_) ? ov_w[(size_t)col*H_ + k] : 0.f);
}

// ---------------- WMMA GEMM: [4096,96]f16 x [96,32000]f16 -> [4096,32000]f32 ----------------
// Operands pre-packed in fragment order: each lane's v16h is one contiguous 32B load.
__global__ void __launch_bounds__(256)
k_gemm(const v16h* __restrict__ Af, const v16h* __restrict__ Bf,
       const float* __restrict__ ov_b, float* __restrict__ out){
    int wave = (int)((blockIdx.x*blockDim.x + threadIdx.x) >> 5);
    int lane = threadIdx.x & 31;
    const int NB = V_/64;                 // 500 n-blocks of 64 cols
    int mt = wave / NB;
    int nb = wave % NB;
    if(mt >= M_/16) return;
    int m0 = mt*16, n0 = nb*64;
    int lrow = lane & 15;
    bool hi = lane >= 16;

    v8f c[4];
    #pragma unroll
    for(int j=0;j<4;++j){
        float bv = ov_b[n0 + j*16 + lrow];
        #pragma unroll
        for(int r=0;r<8;++r) c[j][r] = bv;
    }

    #pragma unroll
    for(int ks=0; ks<3; ++ks){
        v16h a = Af[((size_t)mt*3 + ks)*32 + lane];
        #pragma unroll
        for(int j=0;j<4;++j){
            int nt16 = nb*4 + j;
            v16h b = Bf[((size_t)nt16*3 + ks)*32 + lane];
            c[j] = __builtin_amdgcn_wmma_f32_16x16x32_f16(
                false, a, false, b, (short)0, c[j], false, false);
        }
    }

    #pragma unroll
    for(int j=0;j<4;++j){
        int col = n0 + j*16 + lrow;
        #pragma unroll
        for(int r=0;r<8;++r){
            int row = m0 + r + (hi ? 8 : 0);
            out[(size_t)row*V_ + col] = c[j][r];
        }
    }
}

extern "C" void kernel_launch(void* const* d_in, const int* in_sizes, int n_in,
                              void* d_out, int out_size, void* d_ws, size_t ws_size,
                              hipStream_t stream) {
    (void)in_sizes; (void)n_in; (void)out_size; (void)ws_size;
    const int*   ids  = (const int*)  d_in[0];
    const float* h0   = (const float*)d_in[1];
    const float* c0   = (const float*)d_in[2];
    const float* emb  = (const float*)d_in[3];
    const float* w_ih = (const float*)d_in[4];
    const float* w_hh = (const float*)d_in[5];
    const float* b_ih = (const float*)d_in[6];
    const float* b_hh = (const float*)d_in[7];
    const float* u_w  = (const float*)d_in[8];
    const float* u_b  = (const float*)d_in[9];
    const float* w_w  = (const float*)d_in[10];
    const float* w_b  = (const float*)d_in[11];
    const float* v_w  = (const float*)d_in[12];
    const float* v_b  = (const float*)d_in[13];
    const float* oz_w = (const float*)d_in[14];
    const float* oz_b = (const float*)d_in[15];
    const float* oh_w = (const float*)d_in[16];
    const float* oh_b = (const float*)d_in[17];
    const float* ov_w = (const float*)d_in[18];
    const float* ov_b = (const float*)d_in[19];

    float* ws = (float*)d_ws;
    const size_t MH = (size_t)M_*H_;          // 266240
    float* xe    = ws;                        // [B,T,E]
    float* enc   = xe    + MH;                // [B,T,H]
    float* keys  = enc   + MH;
    float* qrs   = keys  + MH;
    float* s_on  = qrs   + MH;                // [B,T] (becomes e_on)
    float* s_off = s_on  + M_;                // [B,T] (becomes e_off)
    float* S_on  = s_off + M_;
    float* S_off = S_on  + M_;
    float* P_on  = S_off + M_;                // [B,T,H]
    float* P_off = P_on  + MH;
    float* ctx   = P_off + MH;
    _Float16* zpf = (_Float16*)(ctx + MH);             // fragment-order A, 32B-aligned
    _Float16* ovf = (_Float16*)(ctx + MH + (M_*KP)/2); // fragment-order B, 32B-aligned

    float* out = (float*)d_out;
    float* hT  = out + (size_t)M_*V_;
    float* cT  = hT + B_*H_;

    k_embed <<<(M_*E_+255)/256, 256, 0, stream>>>(ids, emb, xe);
    k_lstm  <<<1, 512, 0, stream>>>(xe, w_ih, w_hh, b_ih, b_hh, h0, c0, enc, hT, cT);
    k_kq    <<<(M_*H_+255)/256, 256, 0, stream>>>(enc, u_w, u_b, w_w, w_b, keys, qrs);
    k_scores<<<(M_+255)/256, 256, 0, stream>>>(keys, qrs, v_w, v_b, s_on, s_off);
    k_smax  <<<B_, T_, 0, stream>>>(s_on, s_off, S_on, S_off);
    k_prefix<<<(B_*H_+63)/64, 64, 0, stream>>>(s_on, s_off, qrs, P_on, P_off);
    k_ctx   <<<(M_*H_+255)/256, 256, 0, stream>>>(P_on, P_off, S_on, S_off, ctx);
    k_zf    <<<(M_*KP+255)/256, 256, 0, stream>>>(enc, ctx, oh_w, oh_b, oz_w, oz_b, zpf);
    k_ovf   <<<(V_*KP+255)/256, 256, 0, stream>>>(ov_w, ovf);
    k_gemm  <<<(M_/16)*(V_/64)/8, 256, 0, stream>>>((const v16h*)zpf, (const v16h*)ovf, ov_b, out);
}
